// Wavenet_80066780332492
// MI455X (gfx1250) — matrix operand order repaired and checked by reading, not compiled, not run on hardware
//
#include <hip/hip_runtime.h>
#include <hip/hip_bf16.h>
#include <math.h>
#include <stdint.h>

// ---------------------------------------------------------------------------
// WaveNet forward for MI455X (gfx1250, wave32, WMMA).
// B=4 T=8192 C=128 R=64 S=256 SO=256 O=256 L=16, DIL[i]=2^(i%9)
// bf16 WMMA (f32 accum) everywhere; f32 residual chain; aux outputs
// (in_acts, cond) stream to d_out and are re-read for the fused skip GEMM.
// LDS B-operands staged TRANSPOSED ([N][K]) -> 2x ds_load_b128 per fragment.
// Head-GEMM staging uses CDNA5 async global->LDS copies (ASYNCcnt).
// ---------------------------------------------------------------------------

typedef __bf16 bf16_t;
typedef __attribute__((ext_vector_type(16))) __bf16 v16bf;
typedef __attribute__((ext_vector_type(8)))  __bf16 v8bf;
typedef __attribute__((ext_vector_type(8)))  float  v8f;

#define NB   4
#define NT   8192
#define NL   16
#define SQRT_HALF 0.70710678118654752440f

union Frag16 { v16bf v; v8bf h[2]; };

// ----- WMMA fragment helpers (wave32 layouts per CDNA5 ISA 7.12.2) ---------

// A fragment: 16x32 bf16, rows [row0,row0+16), cols [k0,k0+32) of row-major W
// (ld = ldw).  Per lane: two contiguous 8xbf16 (16B) runs.
__device__ __forceinline__ v16bf load_a_frag(const bf16_t* W, int row0, int ldw,
                                             int k0, int lane) {
  int half = lane >> 4, l16 = lane & 15;
  const bf16_t* p = W + (size_t)(row0 + l16) * ldw + k0 + half * 8;
  Frag16 f;
  f.h[0] = *(const v8bf*)(p);
  f.h[1] = *(const v8bf*)(p + 16);
  return f.v;
}

// B fragment from TRANSPOSED storage Xt[N][K] (ld = ldk): two contiguous 8xbf16 (16B) runs.
__device__ __forceinline__ v16bf load_b_fragT(const bf16_t* Xt, int n0, int ldk,
                                              int k0, int lane) {
  int half = lane >> 4, l16 = lane & 15;
  const bf16_t* p = Xt + (size_t)(n0 + l16) * ldk + k0 + half * 8;
  Frag16 f;
  f.h[0] = *(const v8bf*)(p);
  f.h[1] = *(const v8bf*)(p + 16);
  return f.v;
}

__device__ __forceinline__ v8f wmma_bf16(v16bf a, v16bf b, v8f c) {
  return __builtin_amdgcn_wmma_f32_16x16x32_bf16(false, a, false, b,
                                                 (short)0, c, false, false);
}

// D/C layout: element r of v8f: M = rowTile + r + 8*half, N = colTile + lane%16.

__device__ __forceinline__ float sigmoidf_(float x) {
  return 1.0f / (1.0f + expf(-x));
}

// CDNA5 async global->LDS copy (ISA 15.18.3 op 98), tracked by ASYNCcnt.
// ldsOff = byte offset within the workgroup LDS allocation (low 32 bits of the
// generic LDS address, per ISA 10.2 aperture mapping); gaddr = global address.
__device__ __forceinline__ void async_g2l_b128(uint32_t ldsOff, uint64_t gaddr) {
  asm volatile("global_load_async_to_lds_b128 %0, %1, off"
               :: "v"(ldsOff), "v"(gaddr) : "memory");
}
__device__ __forceinline__ void wait_asynccnt0() {
  asm volatile("s_wait_asynccnt 0x0" ::: "memory");
}

// ---------------------------------------------------------------------------
// Kernel 0: convert / rearrange weights to bf16 in workspace.
// ---------------------------------------------------------------------------
__global__ void convert_weights_kernel(
    const float* cond_W, const float* dilate_W, const float* res_W,
    const float* skip_W, const float* skip_b, const float* conv_out_W,
    const float* conv_end_W,
    bf16_t* condWb, bf16_t* W0b, bf16_t* W1b, bf16_t* resWb,
    bf16_t* skipWb, float* skipBsum, bf16_t* convOutWb, bf16_t* convEndWb) {
  int stride = gridDim.x * blockDim.x;
  int tid = blockIdx.x * blockDim.x + threadIdx.x;
  for (int i = tid; i < 2048 * 128; i += stride) condWb[i] = (bf16_t)cond_W[i];
  for (int i = tid; i < NL * 128 * 64; i += stride) {
    W0b[i] = (bf16_t)dilate_W[2 * i + 0];  // tap k=0 (x[t-d])
    W1b[i] = (bf16_t)dilate_W[2 * i + 1];  // tap k=1 (x[t])
  }
  for (int i = tid; i < 15 * 64 * 64; i += stride) resWb[i] = (bf16_t)res_W[i];
  // fused skip weight: row o (0..255), col = layer*64 + c  -> 256 x 1024
  for (int i = tid; i < 256 * 1024; i += stride) {
    int o = i >> 10, col = i & 1023, l = col >> 6, c = col & 63;
    skipWb[i] = (bf16_t)skip_W[((size_t)l * 256 + o) * 64 + c];
  }
  for (int i = tid; i < 256 * 256; i += stride) {
    convOutWb[i] = (bf16_t)conv_out_W[i];
    convEndWb[i] = (bf16_t)conv_end_W[i];
  }
  for (int i = tid; i < 256; i += stride) {
    float s = 0.f;
    for (int l = 0; l < NL; ++l) s += skip_b[l * 256 + i];
    skipBsum[i] = s;
  }
}

// ---------------------------------------------------------------------------
// Kernel 1: embedding gather -> x0 (f32)   x[b][r][t] = embed[tok[b][t]][r]
// ---------------------------------------------------------------------------
__global__ void embed_kernel(const int* tokens, const float* embedW, float* x0) {
  int idx = blockIdx.x * blockDim.x + threadIdx.x;
  if (idx >= NB * 64 * NT) return;
  int t = idx & (NT - 1);
  int r = (idx >> 13) & 63;
  int b = idx >> 19;
  int tok = tokens[b * NT + t];
  x0[idx] = embedW[tok * 64 + r];
}

// ---------------------------------------------------------------------------
// Kernel 2: cond GEMM: cond[b] (2048 x T) = cond_W (2048x128) @ features[b]
// 128x128 output tile/WG; features staged transposed: XsT[t][c], ld=136.
// ---------------------------------------------------------------------------
__global__ void cond_kernel(const float* features, const bf16_t* condWb,
                            const float* cond_b, float* condOut) {
  __shared__ __attribute__((aligned(16))) bf16_t XsT[128 * 136];
  int bx = blockIdx.x;            // M block (16 of them)
  int t0 = blockIdx.y * 128;      // time tile
  int b  = blockIdx.z;
  int tid = threadIdx.x;
  for (int i = tid; i < 128 * 128; i += 256) {
    int j = i & 127, c = i >> 7;   // consecutive threads: consecutive t
    XsT[j * 136 + c] = (bf16_t)features[((size_t)b * 128 + c) * NT + t0 + j];
  }
  __syncthreads();
  int lane = tid & 31, w = tid >> 5, half = lane >> 4, l16 = lane & 15;
  int row0 = bx * 128 + w * 16;
  v8f acc[8] = {};
#pragma unroll
  for (int kk = 0; kk < 128; kk += 32) {
    __builtin_prefetch(condWb + (size_t)row0 * 128 + kk + 32, 0, 3);
    v16bf a = load_a_frag(condWb, row0, 128, kk, lane);
#pragma unroll
    for (int n = 0; n < 8; ++n) {
      v16bf bb = load_b_fragT(XsT, n * 16, 136, kk, lane);
      acc[n] = wmma_bf16(a, bb, acc[n]);
    }
  }
#pragma unroll
  for (int n = 0; n < 8; ++n)
#pragma unroll
    for (int r = 0; r < 8; ++r) {
      int o = row0 + r + 8 * half;
      int t = t0 + n * 16 + l16;
      condOut[((size_t)b * 2048 + o) * NT + t] = acc[n][r] + cond_b[o];
    }
}

// ---------------------------------------------------------------------------
// Kernel 3 (x16): one WaveNet layer over a 128-wide time tile.
//   ia = W0 @ x[t-d] + W1 @ x[t] + b        (128 x 128 WMMA)
//   in_acts <- ia ; a = ia + cond ; acts = tanh(a_lo)*sigmoid(a_hi)
//   if hasRes: x_out = (res_W @ acts + res_b + x) * sqrt(0.5)
// LDS: smemA = xprevT[128][72] | xcurT[128][72], aliased later as aT[128][136];
//      actsT[128][72] for the res GEMM B-operand.  Total 55,296 B.
// ---------------------------------------------------------------------------
__global__ void layer_kernel(const float* xin, float* xout,
                             const bf16_t* W0, const bf16_t* W1,
                             const bf16_t* resW, const float* db,
                             const float* rb, const float* condBase,
                             float* inactsBase, int layer, int d, int hasRes) {
  __shared__ __attribute__((aligned(16))) bf16_t smemA[2 * 128 * 72];
  __shared__ __attribute__((aligned(16))) bf16_t actsT[128 * 72];
  bf16_t* xprevT = smemA;            // [t][r], ld 72
  bf16_t* xcurT  = smemA + 128 * 72; // [t][r], ld 72
  bf16_t* aT     = smemA;            // alias after ia GEMM: [t][o], ld 136

  int t0 = blockIdx.x * 128;
  int b  = blockIdx.y;
  int tid = threadIdx.x;

  for (int i = tid; i < 64 * 128; i += 256) {
    int j = i & 127, r = i >> 7;     // consecutive threads: consecutive t
    int t = t0 + j, tp = t - d;
    xcurT[j * 72 + r]  = (bf16_t)xin[((size_t)b * 64 + r) * NT + t];
    xprevT[j * 72 + r] = (tp >= 0) ? (bf16_t)xin[((size_t)b * 64 + r) * NT + tp]
                                   : (bf16_t)0.f;
  }
  __syncthreads();

  int lane = tid & 31, w = tid >> 5, half = lane >> 4, l16 = lane & 15;
  int row0 = w * 16;                 // each wave owns one 16-row M tile
  v8f acc[8] = {};
#pragma unroll
  for (int kk = 0; kk < 64; kk += 32) {
    v16bf a0 = load_a_frag(W0, row0, 64, kk, lane);
#pragma unroll
    for (int n = 0; n < 8; ++n) {
      v16bf bb = load_b_fragT(xprevT, n * 16, 72, kk, lane);
      acc[n] = wmma_bf16(a0, bb, acc[n]);
    }
    v16bf a1 = load_a_frag(W1, row0, 64, kk, lane);
#pragma unroll
    for (int n = 0; n < 8; ++n) {
      v16bf bb = load_b_fragT(xcurT, n * 16, 72, kk, lane);
      acc[n] = wmma_bf16(a1, bb, acc[n]);
    }
  }
  __syncthreads();  // all waves done reading smemA -> reuse it as aT

  const float* cond  = condBase   + ((size_t)(b * NL + layer)) * 128 * NT;
  float*       inact = inactsBase + ((size_t)(b * NL + layer)) * 128 * NT;
#pragma unroll
  for (int n = 0; n < 8; ++n) {
    int col = n * 16 + l16;          // t - t0
    int o0  = row0 + 8 * half;       // 8 consecutive output channels
    v8bf pk;
#pragma unroll
    for (int r = 0; r < 8; ++r) {
      int o = o0 + r;
      float ia = acc[n][r] + db[o];                 // in_acts EXCLUDES cond
      inact[(size_t)o * NT + t0 + col] = ia;
      pk[r] = (bf16_t)(ia + cond[(size_t)o * NT + t0 + col]);
    }
    *(v8bf*)&aT[(size_t)col * 136 + o0] = pk;       // one 16B ds_store_b128
  }
  __syncthreads();

  for (int i = tid; i < 64 * 128; i += 256) {       // gated activation
    int r = i & 63, j = i >> 6;      // contiguous LDS reads/writes
    float ta = tanhf((float)aT[j * 136 + r]);
    float sg = sigmoidf_((float)aT[j * 136 + 64 + r]);
    actsT[j * 72 + r] = (bf16_t)(ta * sg);
  }
  __syncthreads();

  if (hasRes) {                                     // 64x64 res GEMM + x update
#pragma unroll
    for (int q = 0; q < 4; ++q) {
      int tt = w * 4 + q, mt = tt >> 3, nt = tt & 7;
      v8f c2 = {};
#pragma unroll
      for (int kk = 0; kk < 64; kk += 32) {
        v16bf a  = load_a_frag(resW, mt * 16, 64, kk, lane);
        v16bf bb = load_b_fragT(actsT, nt * 16, 72, kk, lane);
        c2 = wmma_bf16(a, bb, c2);
      }
#pragma unroll
      for (int r = 0; r < 8; ++r) {
        int m = mt * 16 + r + 8 * half;
        int t = t0 + nt * 16 + l16;
        float xv = xin[((size_t)b * 64 + m) * NT + t];
        xout[((size_t)b * 64 + m) * NT + t] = (c2[r] + rb[m] + xv) * SQRT_HALF;
      }
    }
  }
  // last layer: x is dead (reference reuses previous res_acts; skip path
  // re-derives acts from in_acts+cond in skip_kernel) -> nothing to do.
}

// ---------------------------------------------------------------------------
// Kernel 4: fused skip GEMM over all 16 layers.
//   ht[b][t][:] = relu( sum_i skip_W[i] @ acts_i + sum_i skip_b[i] )
// acts_i regenerated tile-wise from d_out's in_acts + cond regions.
// Output is TIME-MAJOR so the D fragment's 8 consecutive channels become one
// packed 16B store, and downstream staging is a contiguous block copy.
// ---------------------------------------------------------------------------
__global__ void skip_kernel(const float* inactsBase, const float* condBase,
                            const bf16_t* skipWb, const float* skipBsum,
                            bf16_t* ht) {
  __shared__ __attribute__((aligned(16))) bf16_t actsT[128 * 72]; // [t][c]
  int t0 = blockIdx.x * 128, b = blockIdx.y, tid = threadIdx.x;
  int lane = tid & 31, w = tid >> 5, half = lane >> 4, l16 = lane & 15;
  v8f acc[2][8] = {};
  for (int li = 0; li < NL; ++li) {
    const float* ia = inactsBase + ((size_t)(b * NL + li)) * 128 * NT;
    const float* cd = condBase   + ((size_t)(b * NL + li)) * 128 * NT;
    for (int i = tid; i < 64 * 128; i += 256) {
      int j = i & 127, r = i >> 7;   // coalesced global reads along t
      size_t t = (size_t)t0 + j;
      float a0 = ia[(size_t)r * NT + t]        + cd[(size_t)r * NT + t];
      float a1 = ia[(size_t)(r + 64) * NT + t] + cd[(size_t)(r + 64) * NT + t];
      actsT[j * 72 + r] = (bf16_t)(tanhf(a0) * sigmoidf_(a1));
    }
    __syncthreads();
#pragma unroll
    for (int mt = 0; mt < 2; ++mt) {
      int row0 = (w * 2 + mt) * 16;
#pragma unroll
      for (int kk = 0; kk < 64; kk += 32) {
        __builtin_prefetch(skipWb + (size_t)row0 * 1024 + li * 64 + kk + 64, 0, 3);
        v16bf a = load_a_frag(skipWb, row0, 1024, li * 64 + kk, lane);
#pragma unroll
        for (int n = 0; n < 8; ++n) {
          v16bf bb = load_b_fragT(actsT, n * 16, 72, kk, lane);
          acc[mt][n] = wmma_bf16(a, bb, acc[mt][n]);
        }
      }
    }
    __syncthreads();  // before actsT is overwritten by next layer
  }
#pragma unroll
  for (int mt = 0; mt < 2; ++mt)
#pragma unroll
    for (int n = 0; n < 8; ++n) {
      int m0 = (w * 2 + mt) * 16 + 8 * half;   // 8 consecutive channels
      int t  = t0 + n * 16 + l16;
      v8bf pk;
#pragma unroll
      for (int r = 0; r < 8; ++r)
        pk[r] = (bf16_t)fmaxf(acc[mt][n][r] + skipBsum[m0 + r], 0.f);
      *(v8bf*)&ht[((size_t)b * NT + t) * 256 + m0] = pk;  // one 16B store
    }
}

// ---------------------------------------------------------------------------
// Kernel 5/6: 256x256 head GEMM.  X is TIME-MAJOR [b][t][c]; the LDS stage
// XsT[t][c] (ld=264) is filled with CDNA5 async global->LDS b128 copies.
//   mode 0: Yt = relu(W @ X)   (bf16, time-major)
//   mode 1: out[:, :, t+1] = (W @ X)[:, :, t] ; out[:, :, 0] = 0  (f32)
// ---------------------------------------------------------------------------
__global__ void head_kernel(const bf16_t* Xt, const bf16_t* Wg, bf16_t* Yt,
                            float* Yf, int mode) {
  __shared__ __attribute__((aligned(16))) bf16_t XsT[64 * 264];
  int t0 = blockIdx.x * 64, b = blockIdx.y, tid = threadIdx.x;
  // Async copy: 64 rows x 512B (row = one time step, 256 bf16 channels).
  const bf16_t* src = Xt + ((size_t)b * NT + t0) * 256;
  for (int idx = tid; idx < 64 * 32; idx += 256) {
    int row = idx >> 5, chunk = idx & 31;           // 32 x 16B per row
    uint32_t lds = (uint32_t)(uintptr_t)&XsT[row * 264 + chunk * 8];
    uint64_t ga  = (uint64_t)(uintptr_t)(src + (size_t)row * 256 + chunk * 8);
    async_g2l_b128(lds, ga);
  }
  wait_asynccnt0();
  __syncthreads();
  int lane = tid & 31, w = tid >> 5, half = lane >> 4, l16 = lane & 15;
  v8f acc[2][4] = {};
#pragma unroll
  for (int kk = 0; kk < 256; kk += 32) {
#pragma unroll
    for (int mt = 0; mt < 2; ++mt) {
      int row0 = (w * 2 + mt) * 16;
      __builtin_prefetch(Wg + (size_t)row0 * 256 + kk + 32, 0, 3);
      v16bf a = load_a_frag(Wg, row0, 256, kk, lane);
#pragma unroll
      for (int n = 0; n < 4; ++n) {
        v16bf bb = load_b_fragT(XsT, n * 16, 264, kk, lane);
        acc[mt][n] = wmma_bf16(a, bb, acc[mt][n]);
      }
    }
  }
#pragma unroll
  for (int mt = 0; mt < 2; ++mt)
#pragma unroll
    for (int n = 0; n < 4; ++n) {
      int m0 = (w * 2 + mt) * 16 + 8 * half;
      int t  = t0 + n * 16 + l16;
      if (mode == 0) {
        v8bf pk;
#pragma unroll
        for (int r = 0; r < 8; ++r)
          pk[r] = (bf16_t)fmaxf(acc[mt][n][r], 0.f);
        *(v8bf*)&Yt[((size_t)b * NT + t) * 256 + m0] = pk;  // one 16B store
      } else {
#pragma unroll
        for (int r = 0; r < 8; ++r) {
          int m = m0 + r;
          float v = acc[mt][n][r];
          if (t + 1 < NT) Yf[((size_t)b * 256 + m) * NT + t + 1] = v;
          if (t == 0)     Yf[((size_t)b * 256 + m) * NT] = 0.f;
        }
      }
    }
}

// ---------------------------------------------------------------------------
extern "C" void kernel_launch(void* const* d_in, const int* in_sizes, int n_in,
                              void* d_out, int out_size, void* d_ws,
                              size_t ws_size, hipStream_t stream) {
  const float* features   = (const float*)d_in[0];
  const int*   tokens     = (const int*)  d_in[1];
  const float* embedW     = (const float*)d_in[2];
  const float* cond_W     = (const float*)d_in[3];
  const float* cond_b     = (const float*)d_in[4];
  const float* dilate_W   = (const float*)d_in[5];
  const float* dilate_b   = (const float*)d_in[6];
  const float* res_W      = (const float*)d_in[7];
  const float* res_b      = (const float*)d_in[8];
  const float* skip_W     = (const float*)d_in[9];
  const float* skip_b     = (const float*)d_in[10];
  const float* conv_out_W = (const float*)d_in[11];
  const float* conv_end_W = (const float*)d_in[12];

  // d_out = [ out (B,256,T) | in_acts (B,16,128,T) | cond (B,16,128,T) ] f32
  float* outp   = (float*)d_out;
  float* inacts = outp   + (size_t)NB * 256 * NT;
  float* condp  = inacts + (size_t)NB * NL * 128 * NT;

  char* ws = (char*)d_ws;
  auto alloc = [&](size_t bytes) -> char* {
    char* p = ws;
    ws += (bytes + 255) & ~(size_t)255;
    return p;
  };
  bf16_t* condWb    = (bf16_t*)alloc((size_t)2048 * 128 * 2);
  bf16_t* W0b       = (bf16_t*)alloc((size_t)NL * 128 * 64 * 2);
  bf16_t* W1b       = (bf16_t*)alloc((size_t)NL * 128 * 64 * 2);
  bf16_t* resWb     = (bf16_t*)alloc((size_t)15 * 64 * 64 * 2);
  bf16_t* skipWb    = (bf16_t*)alloc((size_t)256 * 1024 * 2);
  bf16_t* convOutWb = (bf16_t*)alloc((size_t)256 * 256 * 2);
  bf16_t* convEndWb = (bf16_t*)alloc((size_t)256 * 256 * 2);
  float*  skipBsum  = (float*) alloc((size_t)256 * 4);
  float*  xA        = (float*) alloc((size_t)NB * 64 * NT * 4);
  float*  xB        = (float*) alloc((size_t)NB * 64 * NT * 4);
  bf16_t* hbuf      = (bf16_t*)alloc((size_t)NB * 256 * NT * 2);  // [b][t][c]
  bf16_t* h2buf     = (bf16_t*)alloc((size_t)NB * 256 * NT * 2);  // [b][t][c]

  convert_weights_kernel<<<512, 256, 0, stream>>>(
      cond_W, dilate_W, res_W, skip_W, skip_b, conv_out_W, conv_end_W,
      condWb, W0b, W1b, resWb, skipWb, skipBsum, convOutWb, convEndWb);

  embed_kernel<<<(NB * 64 * NT) / 256, 256, 0, stream>>>(tokens, embedW, xA);

  cond_kernel<<<dim3(16, NT / 128, NB), 256, 0, stream>>>(features, condWb,
                                                          cond_b, condp);

  float* xb[2] = {xA, xB};
  for (int i = 0; i < NL; ++i) {
    int d = 1 << (i % 9);
    int hasRes = (i < NL - 1) ? 1 : 0;
    layer_kernel<<<dim3(NT / 128, NB), 256, 0, stream>>>(
        xb[i & 1], xb[(i + 1) & 1],
        W0b + (size_t)i * 128 * 64, W1b + (size_t)i * 128 * 64,
        hasRes ? (resWb + (size_t)i * 64 * 64) : resWb,
        dilate_b + i * 128, hasRes ? (res_b + i * 64) : res_b,
        condp, inacts, i, d, hasRes);
  }

  skip_kernel<<<dim3(NT / 128, NB), 256, 0, stream>>>(inacts, condp, skipWb,
                                                      skipBsum, hbuf);

  head_kernel<<<dim3(NT / 64, NB), 256, 0, stream>>>(hbuf, convOutWb, h2buf,
                                                     nullptr, 0);
  head_kernel<<<dim3(NT / 64, NB), 256, 0, stream>>>(h2buf, convEndWb, nullptr,
                                                     outp, 1);
}
